// PQLinear_72430328479853
// MI455X (gfx1250) — matrix operand order repaired and checked by reading, not compiled, not run on hardware
//
#include <hip/hip_runtime.h>
#include <hip/hip_bf16.h>

// PQ-Linear: decode W (bf16) from codebooks, then bf16 WMMA GEMM:
//   out[8192, 4096] = x[8192, 4096] @ W^T[4096, 4096]
// MI455X / gfx1250: wave32, v_wmma_f32_16x16x32_bf16.
// WG tile 128(M) x 256(N), 8 waves (2x4), each wave 4x4 WMMA tiles (64x64).

typedef __attribute__((ext_vector_type(16))) __bf16 v16bf;
typedef __attribute__((ext_vector_type(8)))  __bf16 v8bf;
typedef __attribute__((ext_vector_type(8)))  float  v8f;

#define IN_F   4096
#define OUT_F  4096
#define NSUB   32
#define NCENT  256
#define DSUB   128
#define MTOT   8192          // BATCH * SEQ
#define KC     64            // K chunk per LDS stage
#define LDS_STRIDE 72        // 64 + 8 pad (bf16 elems) = 144B = 9*16B (keeps b128 align)

// Round-to-nearest-even f32 -> 2x bf16 packed into u32.
static __device__ __forceinline__ unsigned pack_bf16x2(float lo, float hi) {
    union { float f; unsigned u; } a, b;
    a.f = lo; b.f = hi;
    unsigned ra = a.u + 0x7FFFu + ((a.u >> 16) & 1u);
    unsigned rb = b.u + 0x7FFFu + ((b.u >> 16) & 1u);
    return (ra >> 16) | (rb & 0xFFFF0000u);
}

// ---------------------------------------------------------------------------
// Kernel 1: decode W[o][k] (bf16) from codebooks + assignments.
// One block per output feature o; 256 threads, 16 elements each.
// ---------------------------------------------------------------------------
__global__ __launch_bounds__(256) void pq_decode_w_bf16(
    const float* __restrict__ codebooks,   // [NSUB][NCENT][DSUB] f32
    const int*   __restrict__ assignments, // [OUT_F][NSUB] i32
    __bf16*      __restrict__ Wb)          // [OUT_F][IN_F] bf16
{
    const int o = blockIdx.x;
    const int t = threadIdx.x;
    const int k = t * 16;                  // 16 consecutive k's, within one subvector
    const int s = k >> 7;                  // subvector index (k / 128)
    const int a = assignments[o * NSUB + s];

    const float4* src = (const float4*)(codebooks +
        ((size_t)(s * NCENT + a)) * DSUB + (k & (DSUB - 1)));
    float4 f0 = src[0], f1 = src[1], f2 = src[2], f3 = src[3];

    uint4 q0, q1;
    q0.x = pack_bf16x2(f0.x, f0.y); q0.y = pack_bf16x2(f0.z, f0.w);
    q0.z = pack_bf16x2(f1.x, f1.y); q0.w = pack_bf16x2(f1.z, f1.w);
    q1.x = pack_bf16x2(f2.x, f2.y); q1.y = pack_bf16x2(f2.z, f2.w);
    q1.z = pack_bf16x2(f3.x, f3.y); q1.w = pack_bf16x2(f3.z, f3.w);

    uint4* dst = (uint4*)(Wb + (size_t)o * IN_F + k);
    dst[0] = q0; dst[1] = q1;
}

// ---------------------------------------------------------------------------
// Kernel 2: bf16 WMMA GEMM.
// ---------------------------------------------------------------------------
__global__ __launch_bounds__(256) void pq_gemm_bf16(
    const float*  __restrict__ X,   // [MTOT][IN_F]  f32
    const __bf16* __restrict__ Wb,  // [OUT_F][IN_F] bf16
    float*        __restrict__ Out) // [MTOT][OUT_F] f32
{
    __shared__ __bf16 As[128 * LDS_STRIDE];   // 18,432 B
    __shared__ __bf16 Bs[256 * LDS_STRIDE];   // 36,864 B

    const int tid   = threadIdx.x;
    const int lane  = tid & 31;
    const int wave  = tid >> 5;      // 0..7
    const int wm    = wave & 1;      // wave row (M): 2
    const int wn    = wave >> 1;     // wave col (N): 4
    const int lrow  = lane & 15;     // 0..15
    const int lhalf = lane >> 4;     // 0..1

    const int mbase = blockIdx.y * 128;
    const int nbase = blockIdx.x * 256;

    // A-fill: 4 threads per row, 16 f32 each; 2 passes cover 128 rows.
    const int a_row = tid >> 2;            // 0..63
    const int a_col = (tid & 3) * 16;      // 0,16,32,48
    // B-fill: 8 threads per row (16B each); 8 passes cover 256 rows.
    const int b_row = tid >> 3;            // 0..31
    const int b_col = (tid & 7) * 8;       // 0..56 (bf16 elems)

    const float*  Xg = X  + (size_t)mbase * IN_F;
    const __bf16* Wg = Wb + (size_t)nbase * IN_F;

    v8f acc[4][4];
#pragma unroll
    for (int mi = 0; mi < 4; ++mi)
#pragma unroll
        for (int ni = 0; ni < 4; ++ni)
#pragma unroll
            for (int j = 0; j < 8; ++j) acc[mi][ni][j] = 0.0f;

    for (int kb = 0; kb < IN_F; kb += KC) {
        // ---- stage A (f32 -> bf16) into LDS ----
#pragma unroll
        for (int pass = 0; pass < 2; ++pass) {
            const int r = a_row + pass * 64;
            const float4* src = (const float4*)(Xg + (size_t)r * IN_F + kb + a_col);
            float4 f0 = src[0], f1 = src[1], f2 = src[2], f3 = src[3];
            uint4 q0, q1;
            q0.x = pack_bf16x2(f0.x, f0.y); q0.y = pack_bf16x2(f0.z, f0.w);
            q0.z = pack_bf16x2(f1.x, f1.y); q0.w = pack_bf16x2(f1.z, f1.w);
            q1.x = pack_bf16x2(f2.x, f2.y); q1.y = pack_bf16x2(f2.z, f2.w);
            q1.z = pack_bf16x2(f3.x, f3.y); q1.w = pack_bf16x2(f3.z, f3.w);
            uint4* dst = (uint4*)(As + r * LDS_STRIDE + a_col);
            dst[0] = q0; dst[1] = q1;
            if (kb + KC < IN_F)
                __builtin_prefetch((const void*)(Xg + (size_t)r * IN_F + kb + KC + a_col), 0, 1);
        }
        // ---- stage B (bf16 copy) into LDS ----
#pragma unroll
        for (int pass = 0; pass < 8; ++pass) {
            const int r = b_row + pass * 32;
            const uint4 u = *(const uint4*)(Wg + (size_t)r * IN_F + kb + b_col);
            *(uint4*)(Bs + r * LDS_STRIDE + b_col) = u;
        }
        if (kb + KC < IN_F)
            __builtin_prefetch((const void*)(Wg + (size_t)b_row * IN_F + kb + KC + b_col), 0, 1);
        __syncthreads();

        // ---- compute: 2 WMMA K-steps of 32 from this chunk ----
#pragma unroll
        for (int kk = 0; kk < 2; ++kk) {
            // A fragment (16-bit A 16x32 layout):
            //   VGPR0-3: K = lhalf*8 + 0..7 ; VGPR4-7: K = 16 + lhalf*8 + 0..7
            v16bf afrag[4];
#pragma unroll
            for (int mi = 0; mi < 4; ++mi) {
                const int m = wm * 64 + mi * 16 + lrow;
                const __bf16* p = As + m * LDS_STRIDE + kk * 32 + lhalf * 8;
                v8bf lo = *(const v8bf*)p;
                v8bf hi = *(const v8bf*)(p + 16);
                afrag[mi] = __builtin_shufflevector(lo, hi,
                    0,1,2,3,4,5,6,7,8,9,10,11,12,13,14,15);
            }
            // B fragment (B 32x16): lane n = lrow, K = lhalf*8 + 0..7, then 16 + lhalf*8 + 0..7 (same as A).
            // Loaded per-ni so only one B fragment is live at a time.
#pragma unroll
            for (int ni = 0; ni < 4; ++ni) {
                const int n = wn * 64 + ni * 16 + lrow;
                const __bf16* p = Bs + n * LDS_STRIDE + kk * 32 + lhalf * 8;
                v8bf lo = *(const v8bf*)p;
                v8bf hi = *(const v8bf*)(p + 16);
                v16bf bfrag = __builtin_shufflevector(lo, hi,
                    0,1,2,3,4,5,6,7,8,9,10,11,12,13,14,15);
#pragma unroll
                for (int mi = 0; mi < 4; ++mi)
                    acc[mi][ni] = __builtin_amdgcn_wmma_f32_16x16x32_bf16(
                        false, afrag[mi], false, bfrag,
                        (short)0, acc[mi][ni], false, false);
            }
        }
        __syncthreads();
    }

    // ---- store C: VGPR r -> (M = lhalf*8 + r, N = lrow) within each 16x16 tile ----
#pragma unroll
    for (int mi = 0; mi < 4; ++mi) {
#pragma unroll
        for (int r = 0; r < 8; ++r) {
            const int m = mbase + wm * 64 + mi * 16 + lhalf * 8 + r;
#pragma unroll
            for (int ni = 0; ni < 4; ++ni) {
                const int n = nbase + wn * 64 + ni * 16 + lrow;
                Out[(size_t)m * OUT_F + n] = acc[mi][ni][r];
            }
        }
    }
}

// ---------------------------------------------------------------------------
extern "C" void kernel_launch(void* const* d_in, const int* in_sizes, int n_in,
                              void* d_out, int out_size, void* d_ws, size_t ws_size,
                              hipStream_t stream) {
    const float* x           = (const float*)d_in[0]; // [4,2048,4096]
    const float* codebooks   = (const float*)d_in[1]; // [32,256,128]
    const int*   assignments = (const int*)d_in[2];   // [4096,32]
    float*       out         = (float*)d_out;         // [4,2048,4096]

    __bf16* Wb = (__bf16*)d_ws;                       // 4096*4096*2 = 32 MB scratch

    pq_decode_w_bf16<<<dim3(OUT_F), dim3(256), 0, stream>>>(codebooks, assignments, Wb);
    pq_gemm_bf16<<<dim3(OUT_F / 256, MTOT / 128), dim3(256), 0, stream>>>(x, Wb, out);
}
